// GlobalPointerRelModel_8383776162432
// MI455X (gfx1250) — matrix operand order repaired and checked by reading, not compiled, not run on hardware
//
#include <hip/hip_runtime.h>

typedef __attribute__((ext_vector_type(16))) __bf16 v16bf;
typedef __attribute__((ext_vector_type(8)))  __bf16 v8bf;
typedef __attribute__((ext_vector_type(8)))  float  v8f;

#define HS      64
#define SEQ     512
#define BATCH   4
#define HID     1024
#define NHEADS  50                 // 2 ent + 24 head + 24 tail
#define NOUT    (NHEADS * 2 * HS)  // 6400
#define MROWS   (BATCH * SEQ)      // 2048

// ---------- helpers ----------

__device__ __forceinline__ __bf16 f2bf(float f) {
  unsigned u = __builtin_bit_cast(unsigned, f);
  u += 0x7FFFu + ((u >> 16) & 1u);                 // round-to-nearest-even
  unsigned short h = (unsigned short)(u >> 16);
  return __builtin_bit_cast(__bf16, h);
}

// A fragment (16x32 bf16, MxK): rowPtr already includes row*ld + (lane>>4)*16.
// Lane holds K = kOff + {0..15} (shifted by +16 for lanes 16..31).
__device__ __forceinline__ v16bf load_frag_a(const __bf16* __restrict__ rowPtr, int kOff) {
  v8bf lo = *(const v8bf*)(rowPtr + kOff);
  v8bf hi = *(const v8bf*)(rowPtr + kOff + 8);
  v16bf r;
#pragma unroll
  for (int i = 0; i < 8; ++i) { r[i] = lo[i]; r[i + 8] = hi[i]; }
  return r;
}

// B fragment (32x16 bf16, KxN) from a row-major (N, K) buffer:
// p already includes n*ld + (lane>>4)*16; 16 contiguous K values per lane.
__device__ __forceinline__ v16bf load_frag_b(const __bf16* __restrict__ p) {
  v8bf lo = *(const v8bf*)(p);
  v8bf hi = *(const v8bf*)(p + 8);
  v16bf r;
#pragma unroll
  for (int i = 0; i < 8; ++i) { r[i] = lo[i]; r[i + 8] = hi[i]; }
  return r;
}

// ---------- prep kernels ----------

__global__ void k_cvt_x(const float* __restrict__ x, __bf16* __restrict__ xb, int n) {
  int i = blockIdx.x * blockDim.x + threadIdx.x;
  if (i < n) xb[i] = f2bf(x[i]);
}

// Wt[n][k] = W(k, n_local), concatenating W_ent | W_head | W_tail along n.
__global__ void k_cvt_w(const float* __restrict__ We, const float* __restrict__ Wh,
                        const float* __restrict__ Wl, __bf16* __restrict__ Wt, int n) {
  int i = blockIdx.x * blockDim.x + threadIdx.x;
  if (i >= n) return;
  int nn = i / HID, k = i - nn * HID;
  float v;
  if (nn < 256)       v = We[k * 256  + nn];
  else if (nn < 3328) v = Wh[k * 3072 + (nn - 256)];
  else                v = Wl[k * 3072 + (nn - 3328)];
  Wt[i] = f2bf(v);
}

// Per-token cos/sin table: cs[(b*512+t)*64 + j] = {cos, sin}(pos * div[(j&31)>>1])
__global__ void k_rope_tab(const int* __restrict__ xp, const int* __restrict__ yp,
                           float2* __restrict__ cs, int n) {
  int i = blockIdx.x * blockDim.x + threadIdx.x;
  if (i >= n) return;
  int tok = i >> 6, j = i & 63;
  int pos = (j < 32) ? xp[tok] : yp[tok];
  int fi  = (j & 31) >> 1;
  float f = __expf(-(float)fi * 0.5756462732485114f);  // ln(1e4)/16
  float a = (float)pos * f;
  cs[i] = make_float2(__cosf(a), __sinf(a));
}

// ---------- projection GEMM + bias + RoPE -> q/k bf16 ----------

__global__ __launch_bounds__(256) void k_proj(
    const __bf16* __restrict__ xb, const __bf16* __restrict__ Wt,
    const float* __restrict__ bent, const float* __restrict__ bhead,
    const float* __restrict__ btail, const float2* __restrict__ cs,
    __bf16* __restrict__ qbuf, __bf16* __restrict__ kbuf) {
  int wave = (blockIdx.x * blockDim.x + threadIdx.x) >> 5;
  int lane = threadIdx.x & 31;
  const int nGroups = NOUT / 64;          // 100
  int mTile = wave / nGroups;             // 0..127
  int nGrp  = wave - mTile * nGroups;
  int m0 = mTile * 16, n0 = nGrp * 64;
  int half = lane >> 4, lmod = lane & 15;

  const __bf16* aRow = xb + (size_t)(m0 + lmod) * HID + half * 16;
  const __bf16* bRow = Wt + (size_t)(n0 + lmod) * HID + half * 16;

  v8f acc[4] = {};
#pragma unroll 4
  for (int kk = 0; kk < HID; kk += 32) {
    __builtin_prefetch(aRow + kk + 128, 0, 0);
    v16bf a = load_frag_a(aRow, kk);
#pragma unroll
    for (int s = 0; s < 4; ++s) {
      v16bf bm = load_frag_b(bRow + (size_t)s * 16 * HID + kk);
      acc[s] = __builtin_amdgcn_wmma_f32_16x16x32_bf16(
          false, a, false, bm, (short)0, acc[s], false, false);
    }
  }

#pragma unroll
  for (int s = 0; s < 4; ++s) {
    int c = n0 + s * 16 + lmod;           // output feature column
    float bias;
    if (c < 256)       bias = bent[c];
    else if (c < 3328) bias = bhead[c - 256];
    else               bias = btail[c - 3328];
    int head   = c >> 7;                  // global head 0..49
    int within = c & 127;
    int isK    = within >> 6;             // 0 = q, 1 = k
    int j      = within & 63;             // dim within head
    int jodd   = j & 1;
    __bf16* dst = isK ? kbuf : qbuf;
#pragma unroll
    for (int r = 0; r < 8; ++r) {
      int row = m0 + r + half * 8;        // global token row = b*512 + t
      float val = acc[s][r] + bias;
      // RoPE pair partner lives in lane^1 (adjacent column)
      float other = __shfl_xor(val, 1, 32);
      float rot = jodd ? other : -other;  // qw2[2i]=-qw[2i+1], qw2[2i+1]=qw[2i]
      float2 cv = cs[(size_t)row * 64 + j];
      float o = val * cv.x + rot * cv.y;
      size_t idx = (((size_t)(row >> 9) * NHEADS + head) * SEQ + (row & 511)) * HS + j;
      dst[idx] = f2bf(o);
    }
  }
}

// ---------- logits = q . k^T per (b, head), + mask/tril/scale ----------

__global__ __launch_bounds__(256) void k_logits(
    const __bf16* __restrict__ qbuf, const __bf16* __restrict__ kbuf,
    const int* __restrict__ att, float* __restrict__ out) {
  int wave = (blockIdx.x * blockDim.x + threadIdx.x) >> 5;
  int lane = threadIdx.x & 31;
  const int nGroups = SEQ / 64;           // 8
  const int mTiles  = SEQ / 16;           // 32
  int bh   = wave / (mTiles * nGroups);   // 0..199
  int rem  = wave - bh * (mTiles * nGroups);
  int mTile = rem / nGroups;
  int nGrp  = rem - mTile * nGroups;
  int b  = bh / NHEADS;
  int hg = bh - b * NHEADS;
  int m0 = mTile * 16, n0 = nGrp * 64;
  int half = lane >> 4, lmod = lane & 15;

  const __bf16* aRow = qbuf + ((size_t)bh * SEQ + m0 + lmod) * HS + half * 16;
  const __bf16* bRow = kbuf + ((size_t)bh * SEQ + n0 + lmod) * HS + half * 16;

  v8f acc[4] = {};
#pragma unroll
  for (int kk = 0; kk < HS; kk += 32) {
    v16bf a = load_frag_a(aRow, kk);
#pragma unroll
    for (int s = 0; s < 4; ++s) {
      v16bf bm = load_frag_b(bRow + (size_t)s * 16 * HS + kk);
      acc[s] = __builtin_amdgcn_wmma_f32_16x16x32_bf16(
          false, a, false, bm, (short)0, acc[s], false, false);
    }
  }

  const int* attB = att + b * SEQ;
  bool isEnt = (hg < 2);
#pragma unroll
  for (int s = 0; s < 4; ++s) {
    int n = n0 + s * 16 + lmod;
    bool mcol = attB[n] > 0;
#pragma unroll
    for (int r = 0; r < 8; ++r) {
      int m = m0 + r + half * 8;
      float v = acc[s][r];
      if (!(mcol && (attB[m] > 0))) v = -__builtin_inff();
      if (isEnt && m > n) v -= 1.0e12f;   // strict lower-triangular penalty
      v *= 0.125f;                        // 1/sqrt(64)
      out[((size_t)bh * SEQ + m) * SEQ + n] = v;
    }
  }
}

// ---------- launch ----------

extern "C" void kernel_launch(void* const* d_in, const int* in_sizes, int n_in,
                              void* d_out, int out_size, void* d_ws, size_t ws_size,
                              hipStream_t stream) {
  const float* x      = (const float*)d_in[0];
  const int*   att    = (const int*)  d_in[1];
  const int*   xp     = (const int*)  d_in[2];
  const int*   yp     = (const int*)  d_in[3];
  const float* W_ent  = (const float*)d_in[4];
  const float* b_ent  = (const float*)d_in[5];
  const float* W_head = (const float*)d_in[6];
  const float* b_head = (const float*)d_in[7];
  const float* W_tail = (const float*)d_in[8];
  const float* b_tail = (const float*)d_in[9];
  float* out = (float*)d_out;

  char* ws = (char*)d_ws;
  size_t off = 0;
  __bf16* xb   = (__bf16*)(ws + off); off += (size_t)MROWS * HID * 2;          //  4.0 MB
  __bf16* Wt   = (__bf16*)(ws + off); off += (size_t)NOUT  * HID * 2;          // 13.1 MB
  __bf16* qbuf = (__bf16*)(ws + off); off += (size_t)BATCH * NHEADS * SEQ * HS * 2;
  __bf16* kbuf = (__bf16*)(ws + off); off += (size_t)BATCH * NHEADS * SEQ * HS * 2;
  float2* cs   = (float2*)(ws + off); off += (size_t)BATCH * SEQ * HS * sizeof(float2);
  (void)ws_size; (void)in_sizes; (void)n_in; (void)out_size;

  int nX = MROWS * HID;
  k_cvt_x<<<(nX + 255) / 256, 256, 0, stream>>>(x, xb, nX);

  int nW = NOUT * HID;
  k_cvt_w<<<(nW + 255) / 256, 256, 0, stream>>>(W_ent, W_head, W_tail, Wt, nW);

  int nR = BATCH * SEQ * HS;
  k_rope_tab<<<(nR + 255) / 256, 256, 0, stream>>>(xp, yp, cs, nR);

  int wavesB = (MROWS / 16) * (NOUT / 64);              // 12800 waves
  k_proj<<<wavesB / 8, 256, 0, stream>>>(xb, Wt, b_ent, b_head, b_tail, cs, qbuf, kbuf);

  int wavesC = BATCH * NHEADS * (SEQ / 16) * (SEQ / 64); // 51200 waves
  k_logits<<<wavesC / 8, 256, 0, stream>>>(qbuf, kbuf, att, out);
}
